// DenseCRFLoss_41738492183297
// MI455X (gfx1250) — compile-verified, matrix operand order checked by reading, not run on hardware
//
#include <hip/hip_runtime.h>
#include <hip/hip_bf16.h>

typedef __attribute__((ext_vector_type(16))) _Float16 v16h;
typedef __attribute__((ext_vector_type(8)))  _Float16 v8h;
typedef __attribute__((ext_vector_type(8)))  float    v8f;

#define NIMG 8
#define KCH  21
#define HIN  128
#define WIN  128
#define PPX  4096           // 64*64 downsampled pixels
#define ROWH 32             // padded f16 row length (K dim for WMMA)
#define QTILES_PER_WAVE 32  // q-tile strip length per wave
#define NBLOCKS 2048        // 8 img * 256 tp * 8 qchunks / 8 waves-per-block

#define LOG2E 1.44269504088896340736f

// ---------------------------------------------------------------------------
// Kernel 1: features. One thread per (n,p). Builds g/h rows such that
//   g_p . h_q = log2(e) * (f_p.f_q - 0.5|f_p|^2 - 0.5|f_q|^2) = log2(e)*(-0.5*d2)
// (log2e folded into the g side only, so the main kernel can use raw v_exp_f32).
//   g = log2e*[r,g,b,x,y, -0.5*|f|^2, 1, 0...]
//   h =       [r,g,b,x,y, 1, -0.5*|f|^2, 0...]
// ---------------------------------------------------------------------------
__global__ void prep_feats_kernel(const float* __restrict__ images,
                                  _Float16* __restrict__ gbuf,
                                  _Float16* __restrict__ hbuf) {
    int id = blockIdx.x * blockDim.x + threadIdx.x;   // 0 .. NIMG*PPX-1
    int n = id >> 12;
    int p = id & (PPX - 1);
    int i = p >> 6;          // y
    int j = p & 63;          // x

    const float inv_rgb = 1.0f / 15.0f;
    const float inv_xy  = 1.0f / 50.0f;  // SIGMA_XY * SCALE = 50

    size_t ib = (size_t)n * 3 * HIN * WIN + (size_t)(2 * i) * WIN + (size_t)(2 * j);
    float r = images[ib]                 * inv_rgb;   // nearest downsample
    float g = images[ib + HIN * WIN]     * inv_rgb;
    float b = images[ib + 2 * HIN * WIN] * inv_rgb;
    float x = (float)j * inv_xy;
    float y = (float)i * inv_xy;
    float sq = r * r + g * g + b * b + x * x + y * y;
    float nh = -0.5f * sq;

    _Float16* gr = gbuf + (size_t)id * ROWH;
    _Float16* hr = hbuf + (size_t)id * ROWH;
    gr[0] = (_Float16)(r * LOG2E);  hr[0] = (_Float16)r;
    gr[1] = (_Float16)(g * LOG2E);  hr[1] = (_Float16)g;
    gr[2] = (_Float16)(b * LOG2E);  hr[2] = (_Float16)b;
    gr[3] = (_Float16)(x * LOG2E);  hr[3] = (_Float16)x;
    gr[4] = (_Float16)(y * LOG2E);  hr[4] = (_Float16)y;
    gr[5] = (_Float16)(nh * LOG2E); hr[5] = (_Float16)1.0f;
    gr[6] = (_Float16)LOG2E;        hr[6] = (_Float16)nh;
#pragma unroll
    for (int t = 7; t < ROWH; ++t) { gr[t] = (_Float16)0.0f; hr[t] = (_Float16)0.0f; }
}

// ---------------------------------------------------------------------------
// Kernel 2: downsample (2x2 mean) + ROI mask of logS / log1_S into packed f16
// rows av[n*P+p][32] (k in [0,21), zero pad to 32). One thread per (n,k,p).
// ---------------------------------------------------------------------------
__global__ void prep_ab_kernel(const float* __restrict__ logS,
                               const float* __restrict__ log1S,
                               const float* __restrict__ ROIs,
                               _Float16* __restrict__ av,
                               _Float16* __restrict__ bv) {
    int id = blockIdx.x * blockDim.x + threadIdx.x;   // 0 .. NIMG*KCH*PPX-1
    int n   = id / (KCH * PPX);
    int rem = id - n * (KCH * PPX);
    int k   = rem >> 12;         // rem / PPX
    int p   = rem & (PPX - 1);
    int i = p >> 6, j = p & 63;

    size_t src = ((size_t)(n * KCH + k) * HIN + (size_t)(2 * i)) * WIN + (size_t)(2 * j);
    float roi = ROIs[((size_t)n * HIN + (size_t)(2 * i)) * WIN + (size_t)(2 * j)];

    float a = 0.25f * (logS[src] + logS[src + 1] + logS[src + WIN] + logS[src + WIN + 1]);
    float b = 0.25f * (log1S[src] + log1S[src + 1] + log1S[src + WIN] + log1S[src + WIN + 1]);

    size_t row = (size_t)(n * PPX + p) * ROWH;
    av[row + k] = (_Float16)(a * roi);
    bv[row + k] = (_Float16)(b * roi);
    if (k == 0) {
#pragma unroll
        for (int t = KCH; t < ROWH; ++t) { av[row + t] = (_Float16)0.0f; bv[row + t] = (_Float16)0.0f; }
    }
}

// ---------------------------------------------------------------------------
// Kernel 3: main fused kernel. One wave32 owns one 16-row p-block and a strip
// of QTILES_PER_WAVE q-tiles. A operands (gA, aA) loaded once, held in VGPRs;
// per q-tile: 4x global_load_b128, 2x v_wmma_f32_16x16x32_f16, then 8x
// (v_exp_f32 + v_fma_f32) -- no guard code, raw hardware exp2 (underflow->0
// is the desired semantics since exponents are <= 0).
// ---------------------------------------------------------------------------
__global__ void __launch_bounds__(256)
densecrf_tile_kernel(const _Float16* __restrict__ gbuf,
                     const _Float16* __restrict__ hbuf,
                     const _Float16* __restrict__ av,
                     const _Float16* __restrict__ bv,
                     float* __restrict__ partials) {
    const int tid  = threadIdx.x;
    const int wave = tid >> 5;
    const int lane = tid & 31;
    const int m    = lane & 15;
    const int hi   = lane >> 4;

    // waveId -> (n, tp, qchunk): 2048 waves per image = 256 tp * 8 qchunks
    const unsigned waveId = blockIdx.x * 8u + (unsigned)wave;   // < 16384
    const int n  = (int)(waveId >> 11);
    const int tp = (int)((waveId >> 3) & 255u);
    const int qc = (int)(waveId & 7u);

    // 16-bit A-matrix lane layout: lanes 0-15 hold K chunks [0..7],[16..23];
    // lanes 16-31 hold [8..15],[24..31]. Rows are 64B; chunks are 16B aligned.
    const int c0 = hi ? 8 : 0;
    const int c1 = c0 + 16;

    const int rowp = tp * 16 + m;
    const _Float16* gp = gbuf + ((size_t)(n << 12) + rowp) * ROWH;
    const _Float16* ap = av   + ((size_t)(n << 12) + rowp) * ROWH;

    // A operands: resident for the whole strip
    v8h gA0 = *(const v8h*)(gp + c0), gA1 = *(const v8h*)(gp + c1);
    v8h aA0 = *(const v8h*)(ap + c0), aA1 = *(const v8h*)(ap + c1);
    v16h gA = __builtin_shufflevector(gA0, gA1, 0,1,2,3,4,5,6,7,8,9,10,11,12,13,14,15);
    v16h aA = __builtin_shufflevector(aA0, aA1, 0,1,2,3,4,5,6,7,8,9,10,11,12,13,14,15);

    const int rowq0 = qc * (QTILES_PER_WAVE * 16) + m;
    const _Float16* hq = hbuf + ((size_t)(n << 12) + rowq0) * ROWH;
    const _Float16* bq = bv   + ((size_t)(n << 12) + rowq0) * ROWH;

    const v8f z = {};
    float s = 0.0f;

#pragma unroll 2
    for (int it = 0; it < QTILES_PER_WAVE; ++it) {
        v8h hB0 = *(const v8h*)(hq + c0), hB1 = *(const v8h*)(hq + c1);
        v8h bB0 = *(const v8h*)(bq + c0), bB1 = *(const v8h*)(bq + c1);
        v16h hB = __builtin_shufflevector(hB0, hB1, 0,1,2,3,4,5,6,7,8,9,10,11,12,13,14,15);
        v16h bB = __builtin_shufflevector(bB0, bB1, 0,1,2,3,4,5,6,7,8,9,10,11,12,13,14,15);

        // E[p,q] = log2(e) * (-0.5 * d2(p,q));  C[p,q] = sum_k a[k,p] b[k,q]
        v8f E = __builtin_amdgcn_wmma_f32_16x16x32_f16(false, gA, false, hB, (short)0, z, false, false);
        v8f C = __builtin_amdgcn_wmma_f32_16x16x32_f16(false, aA, false, bB, (short)0, z, false, false);

#pragma unroll
        for (int r = 0; r < 8; ++r) {
            float w = __builtin_amdgcn_exp2f(E[r]);   // raw v_exp_f32, no guard
            s = __builtin_fmaf(w, C[r], s);
        }

        hq += (size_t)16 * ROWH;   // next q-tile (16 rows of 32 halves)
        bq += (size_t)16 * ROWH;
    }

    // wave32 reduction, once per strip
#pragma unroll
    for (int off = 16; off > 0; off >>= 1) s += __shfl_down(s, off, 32);

    __shared__ float wsum[8];
    if (lane == 0) wsum[wave] = s;
    __syncthreads();
    if (tid == 0) {
        float t = 0.0f;
#pragma unroll
        for (int i = 0; i < 8; ++i) t += wsum[i];
        partials[blockIdx.x] = t;
    }
}

// ---------------------------------------------------------------------------
// Kernel 4: deterministic final reduction of NBLOCKS partials -> scalar.
// out = WEIGHT * sum / N = 1e-9/8 * sum
// ---------------------------------------------------------------------------
__global__ void __launch_bounds__(256)
finalize_kernel(const float* __restrict__ partials, float* __restrict__ out) {
    const int tid = threadIdx.x;
    float s = 0.0f;
    for (int i = tid; i < NBLOCKS; i += 256) s += partials[i];
    __shared__ float red[256];
    red[tid] = s;
    __syncthreads();
    for (int off = 128; off > 0; off >>= 1) {
        if (tid < off) red[tid] += red[tid + off];
        __syncthreads();
    }
    if (tid == 0) out[0] = red[0] * (1e-9f / 8.0f);
}

extern "C" void kernel_launch(void* const* d_in, const int* in_sizes, int n_in,
                              void* d_out, int out_size, void* d_ws, size_t ws_size,
                              hipStream_t stream) {
    const float* images = (const float*)d_in[0];   // [8,3,128,128]
    const float* logS   = (const float*)d_in[1];   // [8,21,128,128]
    const float* log1S  = (const float*)d_in[2];   // [8,21,128,128]
    const float* ROIs   = (const float*)d_in[3];   // [8,128,128]
    float* out = (float*)d_out;

    // Workspace layout: partials | g | h | av | bv (each N*P*32 f16 = 2 MB)
    char* ws = (char*)d_ws;
    float*    partials = (float*)ws;
    _Float16* gbuf = (_Float16*)(ws + (size_t)4096 * sizeof(float));
    _Float16* hbuf = gbuf + (size_t)NIMG * PPX * ROWH;
    _Float16* avb  = hbuf + (size_t)NIMG * PPX * ROWH;
    _Float16* bvb  = avb  + (size_t)NIMG * PPX * ROWH;

    prep_feats_kernel<<<(NIMG * PPX) / 256, 256, 0, stream>>>(images, gbuf, hbuf);
    prep_ab_kernel<<<(NIMG * KCH * PPX) / 256, 256, 0, stream>>>(logS, log1S, ROIs, avb, bvb);
    densecrf_tile_kernel<<<NBLOCKS, 256, 0, stream>>>(gbuf, hbuf, avb, bvb, partials);
    finalize_kernel<<<1, 256, 0, stream>>>(partials, out);
}
